// GAT_43928925503542
// MI455X (gfx1250) — matrix operand order repaired and checked by reading, not compiled, not run on hardware
//
#include <hip/hip_runtime.h>

// ---------------------------------------------------------------------------
// Types / helpers
// ---------------------------------------------------------------------------
typedef __attribute__((ext_vector_type(16))) __bf16 v16bf;
typedef __attribute__((ext_vector_type(8)))  float  v8f;
typedef __attribute__((ext_vector_type(4)))  int    v4i;

union FragBF16 { v16bf v; unsigned int u[8]; };

__device__ __forceinline__ unsigned short f32_to_bf16(float f) {
    union { float f; unsigned int u; } c; c.f = f;
    unsigned int u = c.u;
    unsigned int r = (u + 0x7FFFu + ((u >> 16) & 1u)) >> 16;   // RNE
    return (unsigned short)r;
}
__device__ __forceinline__ float bf16_to_f32(unsigned short h) {
    union { unsigned int u; float f; } c; c.u = ((unsigned int)h) << 16;
    return c.f;
}
__device__ __forceinline__ void atomicMaxF(float* addr, float v) {
    int* ai = (int*)addr;
    int old = *ai;
    while (__int_as_float(old) < v) {
        int prev = atomicCAS(ai, old, __float_as_int(v));
        if (prev == old) break;
        old = prev;
    }
}

// Async global->LDS copy (CDNA5 GLOBAL_LOAD_ASYNC_TO_LDS_B128, ASYNCcnt).
// Builtin signature (from compiler diagnostic): arg0 = AS1 int4*, then LDS ptr.
__device__ __forceinline__ void async_b128(const unsigned int* g, unsigned int* l) {
    __builtin_amdgcn_global_load_async_to_lds_b128(
        (__attribute__((address_space(1))) v4i*)(g),
        (__attribute__((address_space(3))) v4i*)(l), 0, 0);
}
#define WAIT_ASYNCCNT(n) asm volatile("s_wait_asynccnt " #n ::: "memory")

// Problem constants (from the reference)
#define N_NODES 8192
#define NE_RAW  49152
#define ET      57344          // NE_RAW + N_NODES self loops
#define IN_F    128
#define H1      64
#define HID     64
#define F1      4096           // H1*HID
#define H2      5
#define OUT_F   32
#define F2      160            // H2*OUT_F
#define N_BONDS 64

__device__ __forceinline__ void edge_nodes(int e, const int* ei, int& s, int& d) {
    if (e < NE_RAW) { s = ei[e]; d = ei[NE_RAW + e]; }
    else            { s = e - NE_RAW; d = s; }
}

// ---------------------------------------------------------------------------
// Elementwise utility kernels
// ---------------------------------------------------------------------------
__global__ void gat_f32_to_bf16_k(const float* __restrict__ in,
                                  unsigned short* __restrict__ out, int n) {
    int i = blockIdx.x * 256 + threadIdx.x;
    if (i < n) out[i] = f32_to_bf16(in[i]);
}

__global__ void gat_fill_f32_k(float* __restrict__ p, float v, int n) {
    int i = blockIdx.x * 256 + threadIdx.x;
    if (i < n) p[i] = v;
}

// ---------------------------------------------------------------------------
// GEMM1: h1[8192,4096] = x_bf16[8192,128] @ W1_bf16[4096,128]^T  (bf16 WMMA)
// Block: 256 threads (8 waves), tile 64(M) x 128(N), full K=128 in LDS.
// ---------------------------------------------------------------------------
__global__ __launch_bounds__(256)
void gat_gemm1_k(const unsigned short* __restrict__ xb,     // [8192][128]
                 const unsigned short* __restrict__ w1b,    // [4096][128]  ([n][k])
                 unsigned short* __restrict__ h1b) {        // [8192][4096]
    __shared__ unsigned int As[64][65];    // M x Kpairs (64 pairs) + pad
    __shared__ unsigned int Bs[128][65];   // N x Kpairs

    const int t  = threadIdx.x;
    const int m0 = blockIdx.y * 64;
    const int n0 = blockIdx.x * 128;

    const unsigned int* xg = (const unsigned int*)xb;   // pair view, row stride 64
    const unsigned int* wg = (const unsigned int*)w1b;  // pair view, row stride 64

    #pragma unroll
    for (int i = 0; i < 16; ++i) {
        int idx = i * 256 + t; int r = idx >> 6, c = idx & 63;
        As[r][c] = xg[(m0 + r) * 64 + c];
    }
    #pragma unroll
    for (int i = 0; i < 32; ++i) {
        int idx = i * 256 + t; int r = idx >> 6, c = idx & 63;
        Bs[r][c] = wg[(n0 + r) * 64 + c];
    }
    __syncthreads();

    const int lane = t & 31, half = lane >> 4, l16 = lane & 15;
    const int wave = t >> 5;
    const int mW = (wave & 3) * 16;
    const int nW = (wave >> 2) * 64;

    v8f acc[4] = {};
    #pragma unroll
    for (int ks = 0; ks < 4; ++ks) {
        FragBF16 af;
        #pragma unroll
        for (int j = 0; j < 8; ++j) {
            int p = ks * 16 + ((j < 4) ? j : j + 4) + half * 4;   // ISA A-layout
            af.u[j] = As[mW + l16][p];
        }
        #pragma unroll
        for (int nt = 0; nt < 4; ++nt) {
            FragBF16 bf;
            #pragma unroll
            for (int j = 0; j < 8; ++j) {
                int p = ks * 16 + ((j < 4) ? j : j + 4) + half * 4;   // ISA B-layout
                bf.u[j] = Bs[nW + nt * 16 + l16][p];
            }
            acc[nt] = __builtin_amdgcn_wmma_f32_16x16x32_bf16(
                false, af.v, false, bf.v, (short)0, acc[nt], false, false);
        }
    }

    #pragma unroll
    for (int nt = 0; nt < 4; ++nt) {
        #pragma unroll
        for (int i = 0; i < 8; ++i) {
            int m = m0 + mW + i + half * 8;
            int n = n0 + nW + nt * 16 + l16;
            h1b[m * F1 + n] = f32_to_bf16(acc[nt][i]);
        }
    }
}

// ---------------------------------------------------------------------------
// GEMM2: h2[8192,160] = x1_bf16[8192,4096] @ W2_bf16[160,4096]^T (f32 out)
// Block: 256 threads (8 waves), tile 128(M) x 160(N), K step 32, double-
// buffered LDS fed by GLOBAL_LOAD_ASYNC_TO_LDS_B128 (ASYNCcnt pipelining).
// Per-thread async issue per step: 2 A-chunks + 2 B-chunks (+1 B for t<128),
// wave-uniform, so the s_wait_asynccnt immediates are wave-uniform too.
// ---------------------------------------------------------------------------
__global__ __launch_bounds__(256)
void gat_gemm2_k(const unsigned short* __restrict__ x1b,    // [8192][4096]
                 const unsigned short* __restrict__ w2b,    // [160][4096]  ([n][k])
                 float* __restrict__ h2) {                  // [8192][160]
    __shared__ unsigned int As[2][128][17];   // M x 16 Kpairs + pad
    __shared__ unsigned int Bs[2][160][17];

    const int t  = threadIdx.x;
    const int m0 = blockIdx.x * 128;

    const unsigned int* xg = (const unsigned int*)x1b;   // row stride 2048 pairs
    const unsigned int* wg = (const unsigned int*)w2b;

    const int lane = t & 31, half = lane >> 4, l16 = lane & 15;
    const int wave = t >> 5;
    const int mW = wave * 16;

    // Issue one K-tile's async loads into buffer `buf` (kp = K/2 pair offset).
    auto issue_tile = [&](int buf, int kp) {
        // A tile: 128 rows x 4 b128-chunks = 512 chunks, 2 per thread
        #pragma unroll
        for (int i = 0; i < 2; ++i) {
            int c = t + i * 256; int r = c >> 2, c4 = (c & 3) * 4;
            async_b128(&xg[(m0 + r) * 2048 + kp + c4], &As[buf][r][c4]);
        }
        // B tile: 160 rows x 4 chunks = 640 chunks, 2 per thread + 1 for t<128
        #pragma unroll
        for (int i = 0; i < 2; ++i) {
            int c = t + i * 256; int r = c >> 2, c4 = (c & 3) * 4;
            async_b128(&wg[r * 2048 + kp + c4], &Bs[buf][r][c4]);
        }
        if (t < 128) {
            int c = 512 + t; int r = c >> 2, c4 = (c & 3) * 4;
            async_b128(&wg[r * 2048 + kp + c4], &Bs[buf][r][c4]);
        }
    };

    v8f acc[10] = {};

    issue_tile(0, 0);
    const int nsteps = 4096 / 32;               // 128
    for (int st = 0; st < nsteps; ++st) {
        const int buf = st & 1;
        const bool more = (st + 1) < nsteps;
        if (more) issue_tile(buf ^ 1, (st + 1) * 16);

        // Wait for the current buffer's group (older than the just-issued one).
        if (more) {
            if (t < 128) { WAIT_ASYNCCNT(5); }   // wave-uniform branch
            else         { WAIT_ASYNCCNT(4); }
        } else {
            WAIT_ASYNCCNT(0);
        }
        __syncthreads();

        FragBF16 af;
        #pragma unroll
        for (int j = 0; j < 8; ++j) {
            int p = ((j < 4) ? j : j + 4) + half * 4;
            af.u[j] = As[buf][mW + l16][p];
        }
        #pragma unroll
        for (int nt = 0; nt < 10; ++nt) {
            FragBF16 bf;
            #pragma unroll
            for (int j = 0; j < 8; ++j) bf.u[j] = Bs[buf][nt * 16 + l16][((j < 4) ? j : j + 4) + half * 4];
            acc[nt] = __builtin_amdgcn_wmma_f32_16x16x32_bf16(
                false, af.v, false, bf.v, (short)0, acc[nt], false, false);
        }
        __syncthreads();   // protect buf from being re-filled next iteration
    }

    #pragma unroll
    for (int nt = 0; nt < 10; ++nt) {
        #pragma unroll
        for (int i = 0; i < 8; ++i) {
            int m = m0 + mW + i + half * 8;
            int n = nt * 16 + l16;
            h2[m * F2 + n] = acc[nt][i];
        }
    }
}

// ---------------------------------------------------------------------------
// Per-node attention coefficients: alpha_s/alpha_d = <h[n,h,:], a[h,:]>
// ---------------------------------------------------------------------------
__global__ void gat_alpha1_k(const unsigned short* __restrict__ h1b,
                             const float* __restrict__ a_src,
                             const float* __restrict__ a_dst,
                             float* __restrict__ as, float* __restrict__ ad) {
    int idx = blockIdx.x * 256 + threadIdx.x;           // n*H1 + h
    if (idx >= N_NODES * H1) return;
    int n = idx / H1, h = idx - n * H1;
    float s = 0.f, d = 0.f;
    const unsigned short* hp = h1b + n * F1 + h * HID;
    #pragma unroll 8
    for (int c = 0; c < HID; ++c) {
        float v = bf16_to_f32(hp[c]);
        s += v * a_src[h * HID + c];
        d += v * a_dst[h * HID + c];
    }
    as[idx] = s; ad[idx] = d;
}

__global__ void gat_alpha2_k(const float* __restrict__ h2,
                             const float* __restrict__ a_src,
                             const float* __restrict__ a_dst,
                             float* __restrict__ as, float* __restrict__ ad) {
    int idx = blockIdx.x * 256 + threadIdx.x;           // n*H2 + h
    if (idx >= N_NODES * H2) return;
    int n = idx / H2, h = idx - n * H2;
    float s = 0.f, d = 0.f;
    const float* hp = h2 + n * F2 + h * OUT_F;
    #pragma unroll 8
    for (int c = 0; c < OUT_F; ++c) {
        float v = hp[c];
        s += v * a_src[h * OUT_F + c];
        d += v * a_dst[h * OUT_F + c];
    }
    as[idx] = s; ad[idx] = d;
}

// ---------------------------------------------------------------------------
// Edge softmax passes (generic in H)
// ---------------------------------------------------------------------------
__global__ void gat_logits_max_k(const float* __restrict__ as,
                                 const float* __restrict__ ad,
                                 const int* __restrict__ ei, int H,
                                 float* __restrict__ logits,
                                 float* __restrict__ mseg) {
    int idx = blockIdx.x * 256 + threadIdx.x;           // e*H + h
    if (idx >= ET * H) return;
    int e = idx / H, h = idx - e * H;
    int s, d; edge_nodes(e, ei, s, d);
    float l = as[s * H + h] + ad[d * H + h];
    l = (l > 0.f) ? l : 0.2f * l;                       // leaky_relu(0.2)
    logits[idx] = l;
    atomicMaxF(&mseg[d * H + h], l);
}

__global__ void gat_exp_sum_k(const int* __restrict__ ei, int H,
                              float* __restrict__ logits,
                              const float* __restrict__ mseg,
                              float* __restrict__ denom) {
    int idx = blockIdx.x * 256 + threadIdx.x;
    if (idx >= ET * H) return;
    int e = idx / H, h = idx - e * H;
    int s, d; edge_nodes(e, ei, s, d);
    (void)s;
    float v = __expf(logits[idx] - mseg[d * H + h]);
    logits[idx] = v;
    atomicAdd(&denom[d * H + h], v);
}

// Layer-1 aggregation, 4 channels/thread:
// agg[dst, ch..ch+3] += h1[src, ch..ch+3] * alpha[e, ch/HID]
__global__ void gat_agg1_k(const int* __restrict__ ei,
                           const unsigned short* __restrict__ h1b,
                           const float* __restrict__ logits,
                           const float* __restrict__ denom,
                           float* __restrict__ agg) {
    int idx = blockIdx.x * 256 + threadIdx.x;           // e*1024 + q  (q = ch/4)
    int e = idx >> 10, q = idx & 1023;
    if (e >= ET) return;
    int ch = q << 2, h = ch >> 6;                       // 4 | HID => one head
    int s, d; edge_nodes(e, ei, s, d);
    float alpha = logits[e * H1 + h] / denom[d * H1 + h];
    const unsigned int* hp = (const unsigned int*)(h1b + (size_t)s * F1);
    unsigned int p0 = hp[2 * q], p1 = hp[2 * q + 1];    // 4 bf16, 8B aligned
    float* ap = agg + (size_t)d * F1 + ch;
    atomicAdd(ap + 0, bf16_to_f32((unsigned short)(p0 & 0xFFFFu)) * alpha);
    atomicAdd(ap + 1, bf16_to_f32((unsigned short)(p0 >> 16))     * alpha);
    atomicAdd(ap + 2, bf16_to_f32((unsigned short)(p1 & 0xFFFFu)) * alpha);
    atomicAdd(ap + 3, bf16_to_f32((unsigned short)(p1 >> 16))     * alpha);
}

// Layer-2 aggregation, 4 channels/thread.
__global__ void gat_agg2_k(const int* __restrict__ ei,
                           const float* __restrict__ h2,
                           const float* __restrict__ logits,
                           const float* __restrict__ denom,
                           float* __restrict__ agg) {
    int idx = blockIdx.x * 256 + threadIdx.x;           // e*40 + q  (q = ch/4)
    if (idx >= ET * (F2 / 4)) return;
    int e = idx / (F2 / 4), q = idx - e * (F2 / 4);
    int ch = q << 2, h = ch >> 5;                       // 4 | OUT_F => one head
    int s, d; edge_nodes(e, ei, s, d);
    float alpha = logits[e * H2 + h] / denom[d * H2 + h];
    const float4* hp = (const float4*)(h2 + (size_t)s * F2);
    float4 v = hp[q];
    float* ap = agg + (size_t)d * F2 + ch;
    atomicAdd(ap + 0, v.x * alpha);
    atomicAdd(ap + 1, v.y * alpha);
    atomicAdd(ap + 2, v.z * alpha);
    atomicAdd(ap + 3, v.w * alpha);
}

// x1 = bf16(elu(agg1 + b1))
__global__ void gat_x1_k(const float* __restrict__ agg,
                         const float* __restrict__ b1,
                         unsigned short* __restrict__ x1b) {
    int idx = blockIdx.x * 256 + threadIdx.x;
    if (idx >= N_NODES * F1) return;
    float v = agg[idx] + b1[idx & 4095];
    v = (v > 0.f) ? v : (__expf(v) - 1.f);
    x1b[idx] = f32_to_bf16(v);
}

// x2[n,c] = mean_h(agg2[n,h,c]) + b2[c]
__global__ void gat_x2_k(const float* __restrict__ agg2,
                         const float* __restrict__ b2,
                         float* __restrict__ x2) {
    int idx = blockIdx.x * 256 + threadIdx.x;           // n*32 + c
    if (idx >= N_NODES * OUT_F) return;
    int n = idx >> 5, c = idx & 31;
    float s = 0.f;
    #pragma unroll
    for (int h = 0; h < H2; ++h) s += agg2[n * F2 + h * OUT_F + c];
    x2[idx] = s * (1.f / (float)H2) + b2[c];
}

// Bond scores + softmax over 64 bonds
__global__ void gat_score_softmax_k(const float* __restrict__ x2,
                                    const int* __restrict__ lefts,
                                    const int* __restrict__ rights,
                                    float* __restrict__ out) {
    __shared__ float sh[N_BONDS];
    __shared__ float red[2];
    int b = threadIdx.x;
    int L = lefts[b], R = rights[b];
    float s = 0.f;
    #pragma unroll
    for (int c = 0; c < OUT_F; ++c) s += x2[L * OUT_F + c] + x2[R * OUT_F + c];
    sh[b] = s;
    __syncthreads();
    if (b == 0) {
        float mx = -3.0e38f;
        for (int i = 0; i < N_BONDS; ++i) mx = fmaxf(mx, sh[i]);
        float sm = 0.f;
        for (int i = 0; i < N_BONDS; ++i) sm += __expf(sh[i] - mx);
        red[0] = mx; red[1] = sm;
    }
    __syncthreads();
    out[b] = __expf(s - red[0]) / red[1];
}

// ---------------------------------------------------------------------------
// Host launcher
// ---------------------------------------------------------------------------
static inline size_t align_up(size_t v) { return (v + 255) & ~(size_t)255; }

extern "C" void kernel_launch(void* const* d_in, const int* in_sizes, int n_in,
                              void* d_out, int out_size, void* d_ws, size_t ws_size,
                              hipStream_t stream) {
    const float* x        = (const float*)d_in[0];
    const int*   ei       = (const int*)  d_in[1];   // [2][49152] flat
    const int*   lefts    = (const int*)  d_in[2];
    const int*   rights   = (const int*)  d_in[3];
    const float* W1       = (const float*)d_in[4];
    const float* att_src1 = (const float*)d_in[5];
    const float* att_dst1 = (const float*)d_in[6];
    const float* b1       = (const float*)d_in[7];
    const float* W2       = (const float*)d_in[8];
    const float* att_src2 = (const float*)d_in[9];
    const float* att_dst2 = (const float*)d_in[10];
    const float* b2       = (const float*)d_in[11];
    float* out = (float*)d_out;

    // ---- carve workspace ----
    char* base = (char*)d_ws; size_t cur = 0;
    auto take = [&](size_t bytes) { void* p = base + cur; cur += align_up(bytes); return p; };

    unsigned short* xb   = (unsigned short*)take((size_t)N_NODES * IN_F * 2);
    unsigned short* w1b  = (unsigned short*)take((size_t)F1 * IN_F * 2);
    unsigned short* w2b  = (unsigned short*)take((size_t)F2 * F1 * 2);
    unsigned short* h1b  = (unsigned short*)take((size_t)N_NODES * F1 * 2);
    unsigned short* x1b  = (unsigned short*)take((size_t)N_NODES * F1 * 2);
    float* as1    = (float*)take((size_t)N_NODES * H1 * 4);
    float* ad1    = (float*)take((size_t)N_NODES * H1 * 4);
    float* mseg1  = (float*)take((size_t)N_NODES * H1 * 4);
    float* den1   = (float*)take((size_t)N_NODES * H1 * 4);
    float* logits = (float*)take((size_t)ET * H1 * 4);       // reused for layer 2
    float* agg1   = (float*)take((size_t)N_NODES * F1 * 4);  // 128 MB; layer-2 bufs alias it

    // layer-2 buffers carved out of the agg1 region after it is consumed
    char* l2 = (char*)agg1; size_t c2 = 0;
    auto take2 = [&](size_t bytes) { void* p = l2 + c2; c2 += align_up(bytes); return p; };
    float* h2    = (float*)take2((size_t)N_NODES * F2 * 4);
    float* as2   = (float*)take2((size_t)N_NODES * H2 * 4);
    float* ad2   = (float*)take2((size_t)N_NODES * H2 * 4);
    float* mseg2 = (float*)take2((size_t)N_NODES * H2 * 4);
    float* den2  = (float*)take2((size_t)N_NODES * H2 * 4);
    float* agg2  = (float*)take2((size_t)N_NODES * F2 * 4);
    float* x2    = (float*)take2((size_t)N_NODES * OUT_F * 4);
    (void)ws_size; (void)n_in; (void)in_sizes; (void)out_size;

    // ---- conversions to bf16 ----
    {
        int n = N_NODES * IN_F;
        gat_f32_to_bf16_k<<<(n + 255) / 256, 256, 0, stream>>>(x, xb, n);
        n = F1 * IN_F;
        gat_f32_to_bf16_k<<<(n + 255) / 256, 256, 0, stream>>>(W1, w1b, n);
        n = F2 * F1;
        gat_f32_to_bf16_k<<<(n + 255) / 256, 256, 0, stream>>>(W2, w2b, n);
    }

    // ---- layer 1 ----
    gat_gemm1_k<<<dim3(F1 / 128, N_NODES / 64), 256, 0, stream>>>(xb, w1b, h1b);
    gat_alpha1_k<<<(N_NODES * H1) / 256, 256, 0, stream>>>(h1b, att_src1, att_dst1, as1, ad1);

    gat_fill_f32_k<<<(N_NODES * H1) / 256, 256, 0, stream>>>(mseg1, -3.0e38f, N_NODES * H1);
    gat_fill_f32_k<<<(N_NODES * H1) / 256, 256, 0, stream>>>(den1, 0.f, N_NODES * H1);
    gat_fill_f32_k<<<(N_NODES * F1) / 256, 256, 0, stream>>>(agg1, 0.f, N_NODES * F1);

    gat_logits_max_k<<<(ET * H1) / 256, 256, 0, stream>>>(as1, ad1, ei, H1, logits, mseg1);
    gat_exp_sum_k<<<(ET * H1) / 256, 256, 0, stream>>>(ei, H1, logits, mseg1, den1);
    gat_agg1_k<<<(ET * 1024) / 256, 256, 0, stream>>>(ei, h1b, logits, den1, agg1);

    gat_x1_k<<<(N_NODES * F1) / 256, 256, 0, stream>>>(agg1, b1, x1b);

    // ---- layer 2 (buffers alias the now-dead agg1 region) ----
    gat_gemm2_k<<<N_NODES / 128, 256, 0, stream>>>(x1b, w2b, h2);
    gat_alpha2_k<<<(N_NODES * H2 + 255) / 256, 256, 0, stream>>>(h2, att_src2, att_dst2, as2, ad2);

    gat_fill_f32_k<<<(N_NODES * H2 + 255) / 256, 256, 0, stream>>>(mseg2, -3.0e38f, N_NODES * H2);
    gat_fill_f32_k<<<(N_NODES * H2 + 255) / 256, 256, 0, stream>>>(den2, 0.f, N_NODES * H2);
    gat_fill_f32_k<<<(N_NODES * F2 + 255) / 256, 256, 0, stream>>>(agg2, 0.f, N_NODES * F2);

    gat_logits_max_k<<<(ET * H2 + 255) / 256, 256, 0, stream>>>(as2, ad2, ei, H2, logits, mseg2);
    gat_exp_sum_k<<<(ET * H2 + 255) / 256, 256, 0, stream>>>(ei, H2, logits, mseg2, den2);
    gat_agg2_k<<<(ET * (F2 / 4)) / 256, 256, 0, stream>>>(ei, h2, logits, den2, agg2);

    gat_x2_k<<<(N_NODES * OUT_F) / 256, 256, 0, stream>>>(agg2, b2, x2);
    gat_score_softmax_k<<<1, N_BONDS, 0, stream>>>(x2, lefts, rights, out);
}
